// Linear_65558380806418
// MI455X (gfx1250) — matrix operand order repaired and checked by reading, not compiled, not run on hardware
//
#include <hip/hip_runtime.h>

// Linear: x[16,2048,4096] f32 @ w[64,4096]^T + bias[64] -> out[16,2048,64] f32
// M=32768, K=4096, N=64. Memory-bound floor ~22us @ 23.3 TB/s (521 MB traffic).
// Strategy: f32 -> f16 hi/lo split (error ~2^-21) + V_WMMA_F32_16X16X32_F16,
// 3 WMMAs per tile (xh*wh + xl*wh + xh*wl), f32 accumulation.

#define K_DIM 4096
#define N_DIM 64

typedef __attribute__((ext_vector_type(16))) _Float16 v16h;
typedef __attribute__((ext_vector_type(8)))  float    v8f;
typedef __attribute__((ext_vector_type(4)))  float    v4f;

// ---------------------------------------------------------------------------
// Pre-pass: split f32 weight into f16 hi + f16 lo(residual) in workspace.
// 64*4096 = 256K elements; runs once, ~1 MB written, negligible cost.
// ---------------------------------------------------------------------------
__global__ void w_split_kernel(const float* __restrict__ w,
                               _Float16* __restrict__ wh,
                               _Float16* __restrict__ wl, int n) {
    int i = blockIdx.x * blockDim.x + threadIdx.x;
    if (i < n) {
        float f = w[i];
        _Float16 h = (_Float16)f;
        wh[i] = h;
        wl[i] = (_Float16)(f - (float)h);
    }
}

// ---------------------------------------------------------------------------
// Main GEMM. Each wave: 32 M-rows x 64 N-cols. Block = 8 waves = 256 rows.
// Grid = M/256 = 128 blocks.
//
// Fragment layouts used here (wave32; A/B K order is this code's contiguous choice):
//  A (16x32 f16): lane r=lane&15 is row M=r; lanes<16 hold K 0..15,
//                 lanes>=16 hold K 16..31 -> one contiguous 16-f32 run.
//  B (32x16 f16): lane n=lane&15 is col N=n; lanes<16 hold K 0..15,
//                 lanes>=16 hold K 16..31 -> one contiguous 16-f16 (32B) run.
//  C/D (16x16 f32): vgpr v, lanes<16 -> M=v, lanes>=16 -> M=v+8; N=lane&15.
// ---------------------------------------------------------------------------
__global__ __launch_bounds__(256) void linear_wmma_kernel(
    const float* __restrict__ x,
    const _Float16* __restrict__ wh,
    const _Float16* __restrict__ wl,
    const float* __restrict__ bias,
    float* __restrict__ out) {

    const int lane    = threadIdx.x & 31;
    const int wave    = threadIdx.x >> 5;
    const int r       = lane & 15;
    const int halfsel = lane >> 4;               // 0: lanes 0-15, 1: lanes 16-31
    const int koffA   = halfsel * 16;            // A K-offset within 32-chunk
    const int koffB   = halfsel * 16;            // B K-offset within 32-chunk
    const int n_col   = lane & 15;

    const long m0 = (long)blockIdx.x * 256 + (long)wave * 32;  // first of 32 rows

    v8f zero;
#pragma unroll
    for (int i = 0; i < 8; ++i) zero[i] = 0.0f;

    v8f acc[2][4];
#pragma unroll
    for (int mt = 0; mt < 2; ++mt)
#pragma unroll
        for (int nt = 0; nt < 4; ++nt) acc[mt][nt] = zero;

    const float* xrow[2];
    xrow[0] = x + (m0 + r) * K_DIM;
    xrow[1] = x + (m0 + 16 + r) * K_DIM;

    for (int k0 = 0; k0 < K_DIM; k0 += 32) {
        // Prefetch the x stream ~8 iterations ahead into L2/L0.
        __builtin_prefetch(xrow[0] + k0 + 256, 0, 0);
        __builtin_prefetch(xrow[1] + k0 + 256, 0, 0);

        // --- B fragments (already f16 in workspace; contiguous 32B per lane) ---
        v16h bh[4], bl[4];
#pragma unroll
        for (int nt = 0; nt < 4; ++nt) {
            const long boff = (long)(nt * 16 + n_col) * K_DIM + k0 + koffB;
            bh[nt] = *(const v16h*)(wh + boff);
            bl[nt] = *(const v16h*)(wl + boff);
        }

#pragma unroll
        for (int mt = 0; mt < 2; ++mt) {
            // --- A fragment: 16 f32 per lane as 4x b128, split into hi/lo f16 ---
            const float* xr = xrow[mt];
            v4f c0 = *(const v4f*)(xr + k0 + koffA);
            v4f c1 = *(const v4f*)(xr + k0 + koffA + 4);
            v4f c2 = *(const v4f*)(xr + k0 + koffA + 8);
            v4f c3 = *(const v4f*)(xr + k0 + koffA + 12);

            v16h ah, al;
#pragma unroll
            for (int h = 0; h < 4; ++h) {
                float f; _Float16 H;
                f = c0[h]; H = (_Float16)f; ah[h]      = H; al[h]      = (_Float16)(f - (float)H);
                f = c1[h]; H = (_Float16)f; ah[4 + h]  = H; al[4 + h]  = (_Float16)(f - (float)H);
                f = c2[h]; H = (_Float16)f; ah[8 + h]  = H; al[8 + h]  = (_Float16)(f - (float)H);
                f = c3[h]; H = (_Float16)f; ah[12 + h] = H; al[12 + h] = (_Float16)(f - (float)H);
            }

            // --- 3-term hi/lo product per N-tile, f32 accumulate ---
#pragma unroll
            for (int nt = 0; nt < 4; ++nt) {
                acc[mt][nt] = __builtin_amdgcn_wmma_f32_16x16x32_f16(
                    false, ah, false, bh[nt], (short)0, acc[mt][nt], false, false);
                acc[mt][nt] = __builtin_amdgcn_wmma_f32_16x16x32_f16(
                    false, al, false, bh[nt], (short)0, acc[mt][nt], false, false);
                acc[mt][nt] = __builtin_amdgcn_wmma_f32_16x16x32_f16(
                    false, ah, false, bl[nt], (short)0, acc[mt][nt], false, false);
            }
        }
    }

    // --- Epilogue: bias add + store (C/D layout: M = v + 8*halfsel, N = n) ---
#pragma unroll
    for (int nt = 0; nt < 4; ++nt) {
        const int n = nt * 16 + n_col;
        const float b = bias[n];
#pragma unroll
        for (int mt = 0; mt < 2; ++mt) {
            const long mbase = m0 + mt * 16 + halfsel * 8;
#pragma unroll
            for (int v = 0; v < 8; ++v) {
                out[(mbase + v) * N_DIM + n] = acc[mt][nt][v] + b;
            }
        }
    }
}

// ---------------------------------------------------------------------------
extern "C" void kernel_launch(void* const* d_in, const int* in_sizes, int n_in,
                              void* d_out, int out_size, void* d_ws, size_t ws_size,
                              hipStream_t stream) {
    const float* x    = (const float*)d_in[0];   // [16,2048,4096] f32
    const float* w    = (const float*)d_in[1];   // [64,4096] f32
    const float* bias = (const float*)d_in[2];   // [64] f32
    float* out        = (float*)d_out;           // [16,2048,64] f32

    const int M = in_sizes[0] / K_DIM;           // 32768

    _Float16* wh = (_Float16*)d_ws;              // 512 KB
    _Float16* wl = wh + (size_t)N_DIM * K_DIM;   // 512 KB
    const int wn = N_DIM * K_DIM;

    w_split_kernel<<<(wn + 255) / 256, 256, 0, stream>>>(w, wh, wl, wn);

    const int rows_per_block = 256;              // 8 waves x 32 rows
    linear_wmma_kernel<<<M / rows_per_block, 256, 0, stream>>>(x, wh, wl, bias, out);
}
